// RTRLQuasiLSTMlayer_6193342841505
// MI455X (gfx1250) — compile-verified, matrix operand order checked by reading, not run on hardware
//
#include <hip/hip_runtime.h>
#include <math.h>

// RTRL quasi-LSTM: B=64, H=1024, I=1024, all fp32.
// Kernel 1: gate GEMMs via v_wmma_f32_16x16x4_f32 + elementwise epilogue (tiny).
// Kernel 2: bandwidth-bound stream of Z_state/F_state (~1.07 GB => ~46us @ 23.3 TB/s).

typedef __attribute__((ext_vector_type(2)))  float    v2f;
typedef __attribute__((ext_vector_type(4)))  float    v4f;
typedef __attribute__((ext_vector_type(8)))  float    v8f;
typedef __attribute__((ext_vector_type(16))) _Float16 v16h;

#define B_SZ 64
#define H_SZ 1024
#define I_SZ 1024

// ---------------------------------------------------------------------------
// Kernel 1: one wave (32 lanes) computes a 16x16 (b,h) tile of both gate
// pre-activations via WMMA, then the full elementwise epilogue for that tile.
// Grid: (B/16)*(H/16) = 4*64 = 256 blocks of 32 threads (EXEC all-ones).
// ---------------------------------------------------------------------------
__global__ __launch_bounds__(32)
void rtrl_gates_wmma(const float* __restrict__ x,
                     const float* __restrict__ hidden_prev,
                     const float* __restrict__ wz_state,
                     const float* __restrict__ wf_state,
                     const float* __restrict__ bz_state,
                     const float* __restrict__ bf_state,
                     const float* __restrict__ wm_z,
                     const float* __restrict__ wm_f,
                     const float* __restrict__ wv_z,
                     const float* __restrict__ wv_f,
                     const float* __restrict__ bias_z,
                     const float* __restrict__ bias_f,
                     float* __restrict__ new_cell,
                     float* __restrict__ wz_new,
                     float* __restrict__ wf_new,
                     float* __restrict__ bz_new,
                     float* __restrict__ bf_new,
                     float* __restrict__ zf_ws,
                     float* __restrict__ fz_ws,
                     float* __restrict__ cm_ws)
{
    const int lane = threadIdx.x;       // 0..31
    const int lo   = lane & 15;
    const int hi   = lane >> 4;         // 0 or 1
    const int b0   = (blockIdx.x >> 6) << 4;   // 4 b-tiles
    const int h0   = (blockIdx.x & 63) << 4;   // 64 h-tiles

    float accz[8];
    float accf[8];

#if defined(__gfx1250__)
    v8f az = {};
    v8f af = {};
    // A (16x4, M x K): lane holds row M=lo; VGPR comp c -> K = hi*2 + c.
    // B (4x16, K x N): lane holds col N=lo; B[k][n] = wm[h0+n][k] (wm is [H,I] row-major).
    const float* __restrict__ xrow = x    + (size_t)(b0 + lo) * I_SZ;
    const float* __restrict__ zrow = wm_z + (size_t)(h0 + lo) * I_SZ;
    const float* __restrict__ frow = wm_f + (size_t)(h0 + lo) * I_SZ;
#if __has_builtin(__builtin_amdgcn_wmma_f32_16x16x4_f32)
    #pragma unroll 8
    for (int k = 0; k < I_SZ; k += 4) {
        const int ka = k + hi * 2;
        v2f a, bz, bf;
        a.x  = xrow[ka];  a.y  = xrow[ka + 1];
        bz.x = zrow[ka];  bz.y = zrow[ka + 1];
        bf.x = frow[ka];  bf.y = frow[ka + 1];
        az = __builtin_amdgcn_wmma_f32_16x16x4_f32(false, a, false, bz,
                                                   (short)0, az, false, false);
        af = __builtin_amdgcn_wmma_f32_16x16x4_f32(false, a, false, bf,
                                                   (short)0, af, false, false);
    }
#else
    // Fallback: codegen-confirmed f16 WMMA (fp32 accumulate); 16-bit A/B layout
    // per ISA: comps 0..7 -> K = base + hi*8 + j, comps 8..15 -> K + 16.
    for (int k = 0; k < I_SZ; k += 32) {
        v16h a, bz, bf;
        const int k0 = k + hi * 8;
        const int k1 = k + 16 + hi * 8;
        #pragma unroll
        for (int j = 0; j < 8; ++j) {
            a[j]      = (_Float16)xrow[k0 + j];
            a[8 + j]  = (_Float16)xrow[k1 + j];
            bz[j]     = (_Float16)zrow[k0 + j];
            bz[8 + j] = (_Float16)zrow[k1 + j];
            bf[j]     = (_Float16)frow[k0 + j];
            bf[8 + j] = (_Float16)frow[k1 + j];
        }
        az = __builtin_amdgcn_wmma_f32_16x16x32_f16(false, a, false, bz,
                                                    (short)0, az, false, false);
        af = __builtin_amdgcn_wmma_f32_16x16x32_f16(false, a, false, bf,
                                                    (short)0, af, false, false);
    }
#endif
    #pragma unroll
    for (int r = 0; r < 8; ++r) { accz[r] = az[r]; accf[r] = af[r]; }
#else
    // Host-pass / non-gfx1250 stub (never executed on MI455X).
    for (int r = 0; r < 8; ++r) {
        const int b = b0 + hi * 8 + r, h = h0 + lo;
        float sz = 0.f, sf = 0.f;
        for (int k = 0; k < I_SZ; ++k) {
            sz += x[(size_t)b * I_SZ + k] * wm_z[(size_t)h * I_SZ + k];
            sf += x[(size_t)b * I_SZ + k] * wm_f[(size_t)h * I_SZ + k];
        }
        accz[r] = sz; accf[r] = sf;
    }
#endif

    // Epilogue: C/D layout -> lane<16: M=r, lane>=16: M=8+r; N = lane%16.
    #pragma unroll
    for (int r = 0; r < 8; ++r) {
        const int b  = b0 + hi * 8 + r;
        const int h  = h0 + lo;
        const int bh = b * H_SZ + h;
        const float hp  = hidden_prev[bh];
        const float wvz = wv_z[h];
        const float wvf = wv_f[h];
        const float z_part = tanhf(accz[r] + wvz * hp + bias_z[h]);
        const float f_pre  = accf[r] + wvf * hp + bias_f[h];
        const float f_part = 1.0f / (1.0f + expf(-f_pre));
        const float one_mf = 1.0f - f_part;
        const float zf = one_mf * (1.0f - z_part * z_part);
        const float fz = (hp - z_part) * one_mf * f_part;
        const float cm = f_part + zf * wvz + fz * wvf;
        new_cell[bh] = hp * f_part + one_mf * z_part;
        wz_new[bh]   = hp * zf + cm * wz_state[bh];
        wf_new[bh]   = hp * fz + cm * wf_state[bh];
        bz_new[bh]   = zf + cm * bz_state[bh];
        bf_new[bh]   = fz + cm * bf_state[bh];
        zf_ws[bh] = zf;
        fz_ws[bh] = fz;
        cm_ws[bh] = cm;
    }
}

// ---------------------------------------------------------------------------
// Kernel 2: HBM-bandwidth-bound update of the [B,H,I] sensitivity tensors.
// One block per (b,h) row; thread t handles 4 consecutive i (b128 accesses).
// Non-temporal hints on the 1 GB Z/F stream keep x (256 KB, high reuse) in L2.
// ---------------------------------------------------------------------------
__global__ __launch_bounds__(256)
void rtrl_stream_update(const float* __restrict__ Z,
                        const float* __restrict__ F,
                        const float* __restrict__ x,
                        const float* __restrict__ zf_ws,
                        const float* __restrict__ fz_ws,
                        const float* __restrict__ cm_ws,
                        float* __restrict__ Zout,
                        float* __restrict__ Fout)
{
    const int bh = blockIdx.x;          // b*H + h
    const int b  = bh >> 10;            // H = 1024
    const int t  = threadIdx.x;
    const float cm = cm_ws[bh];
    const float zf = zf_ws[bh];
    const float fz = fz_ws[bh];

    const size_t base = (size_t)bh * I_SZ + (size_t)t * 4;
    const v4f xv = *(const v4f*)(x + (size_t)b * I_SZ + (size_t)t * 4);
    const v4f zv = __builtin_nontemporal_load((const v4f*)(Z + base));
    const v4f fv = __builtin_nontemporal_load((const v4f*)(F + base));

    v4f zo, fo;
    zo.x = cm * zv.x + zf * xv.x;
    zo.y = cm * zv.y + zf * xv.y;
    zo.z = cm * zv.z + zf * xv.z;
    zo.w = cm * zv.w + zf * xv.w;
    fo.x = cm * fv.x + fz * xv.x;
    fo.y = cm * fv.y + fz * xv.y;
    fo.z = cm * fv.z + fz * xv.z;
    fo.w = cm * fv.w + fz * xv.w;

    __builtin_nontemporal_store(zo, (v4f*)(Zout + base));
    __builtin_nontemporal_store(fo, (v4f*)(Fout + base));
}

extern "C" void kernel_launch(void* const* d_in, const int* in_sizes, int n_in,
                              void* d_out, int out_size, void* d_ws, size_t ws_size,
                              hipStream_t stream)
{
    (void)in_sizes; (void)n_in; (void)out_size; (void)ws_size;

    const float* x           = (const float*)d_in[0];
    const float* hidden_prev = (const float*)d_in[1];
    const float* Z_state     = (const float*)d_in[2];
    const float* F_state     = (const float*)d_in[3];
    const float* wz_state    = (const float*)d_in[4];
    const float* wf_state    = (const float*)d_in[5];
    const float* bz_state    = (const float*)d_in[6];
    const float* bf_state    = (const float*)d_in[7];
    const float* wm_z        = (const float*)d_in[8];
    const float* wm_f        = (const float*)d_in[9];
    const float* wv_z        = (const float*)d_in[10];
    const float* wv_f        = (const float*)d_in[11];
    const float* bias_z      = (const float*)d_in[12];
    const float* bias_f      = (const float*)d_in[13];

    // Output tuple layout, concatenated flat in return order.
    float* out = (float*)d_out;
    const size_t BH  = (size_t)B_SZ * H_SZ;        // 65536
    const size_t BHI = BH * (size_t)I_SZ;          // 67108864
    float* new_cell = out;
    float* Z_new    = new_cell + BH;
    float* F_new    = Z_new + BHI;
    float* wz_new   = F_new + BHI;
    float* wf_new   = wz_new + BH;
    float* bz_new   = wf_new + BH;
    float* bf_new   = bz_new + BH;

    // Workspace: zf_tmp, fz_tmp, common  (3 * 64K floats = 768 KB).
    float* ws    = (float*)d_ws;
    float* zf_ws = ws;
    float* fz_ws = ws + BH;
    float* cm_ws = ws + 2 * BH;

    rtrl_gates_wmma<<<(B_SZ / 16) * (H_SZ / 16), 32, 0, stream>>>(
        x, hidden_prev, wz_state, wf_state, bz_state, bf_state,
        wm_z, wm_f, wv_z, wv_f, bias_z, bias_f,
        new_cell, wz_new, wf_new, bz_new, bf_new,
        zf_ws, fz_ws, cm_ws);

    rtrl_stream_update<<<B_SZ * H_SZ, 256, 0, stream>>>(
        Z_state, F_state, x, zf_ws, fz_ws, cm_ws, Z_new, F_new);
}
